// ChannelAttentionBlock_42554535968892
// MI455X (gfx1250) — matrix operand order repaired and checked by reading, not compiled, not run on hardware
//
#include <hip/hip_runtime.h>

// Problem constants (from reference): x [8, 128, 64, 64] f32
#define BATCH 8
#define DCH   128
#define NPIX  4096   // 64*64

typedef __attribute__((ext_vector_type(16))) __bf16 v16bf;
typedef __attribute__((ext_vector_type(8)))  __bf16 v8bf;
typedef __attribute__((ext_vector_type(4)))  __bf16 v4bf;
typedef __attribute__((ext_vector_type(2)))  __bf16 v2bf;
typedef __attribute__((ext_vector_type(8)))  float  v8f;

__device__ __forceinline__ __bf16 f2bf(float f) {
  unsigned u = __builtin_bit_cast(unsigned, f);
  u += 0x7FFFu + ((u >> 16) & 1u);               // round-to-nearest-even
  unsigned short h = (unsigned short)(u >> 16);
  return __builtin_bit_cast(__bf16, h);
}

__device__ __forceinline__ v8f wmma_bf16(v16bf a, v16bf b, v8f c) {
  // D = A(16x32 bf16) * B(32x16 bf16) + C(16x16 f32)
  return __builtin_amdgcn_wmma_f32_16x16x32_bf16(false, a, false, b, (short)0, c,
                                                 false, false);
}

// A-fragment: 16 contiguous bf16 at p[0..15] (K run per lane half, matches B).
__device__ __forceinline__ v16bf ldfragA(const __bf16* p) {
  v8bf lo = *(const v8bf*)(p);
  v8bf hi = *(const v8bf*)(p + 8);
  return __builtin_shufflevector(lo, hi, 0,1,2,3,4,5,6,7,8,9,10,11,12,13,14,15);
}

// B-fragment: 16 contiguous bf16 (contiguous K order k = 16h + s per lane half, same as A).
__device__ __forceinline__ v16bf ldfragB(const __bf16* p) {
  v8bf lo = *(const v8bf*)(p);
  v8bf hi = *(const v8bf*)(p + 8);
  return __builtin_shufflevector(lo, hi, 0,1,2,3,4,5,6,7,8,9,10,11,12,13,14,15);
}

// ---------------------------------------------------------------------------
// Pass 0: one-time f32 -> bf16 conversion of F into workspace.
// ---------------------------------------------------------------------------
__global__ __launch_bounds__(256)
void ca_cvt_kernel(const float* __restrict__ x, __bf16* __restrict__ fb) {
  const size_t i = ((size_t)blockIdx.x * 256 + threadIdx.x) * 4;
  const float4 f = *(const float4*)(x + i);
  v4bf o;
  o[0] = f2bf(f.x); o[1] = f2bf(f.y); o[2] = f2bf(f.z); o[3] = f2bf(f.w);
  *(v4bf*)(fb + i) = o;
}

// ---------------------------------------------------------------------------
// Pass 1: per-row softmax stats. S[n, m] = <F[n,:], F[m,:]>; rowmax + sumexp
// over all m (online). Each wave owns 32 rows (two 16-row A tiles held in
// registers) so every streamed B fragment feeds two WMMAs. Workgroup = 8
// waves = 256 rows.
// ---------------------------------------------------------------------------
__global__ __launch_bounds__(256)
void ca_stats_kernel(const __bf16* __restrict__ fb,
                     float* __restrict__ statsM,
                     float* __restrict__ statsS) {
  const int b    = blockIdx.y;
  const int lane = threadIdx.x & 31;
  const int wv   = threadIdx.x >> 5;
  const int r0   = blockIdx.x * 256 + wv * 32;
  const __bf16* F = fb + (size_t)b * NPIX * DCH;

  // A fragments: rows r0..r0+15 (h=0) and r0+16..r0+31 (h=1), K in 4 chunks.
  v16bf afr[2][4];
  {
    const int koff = (lane & 16) ? 16 : 0;
#pragma unroll
    for (int h = 0; h < 2; ++h) {
      const int row = r0 + h * 16 + (lane & 15);
#pragma unroll
      for (int kb = 0; kb < 4; ++kb)
        afr[h][kb] = ldfragA(F + (size_t)row * DCH + kb * 32 + koff);
    }
  }

  float rm[2][8], rs[2][8];
#pragma unroll
  for (int h = 0; h < 2; ++h)
#pragma unroll
    for (int v = 0; v < 8; ++v) { rm[h][v] = -3.0e38f; rs[h][v] = 0.f; }

  const int mlane = lane & 15;
  const int bkoff = (lane & 16) ? 16 : 0;  // B 32x16: lanes<16 K=0..15, else 16..31

  for (int m0 = 0; m0 < NPIX; m0 += 32) {
    v8f acc[2][2];
#pragma unroll
    for (int j = 0; j < 2; ++j) {
      v8f c0 = {};
      v8f c1 = {};
      const int mrow = m0 + j * 16 + mlane;
      // Prefetch same rows of the next 32-wide m block (wraps; speculative).
      const int pfrow = ((m0 + 32) & (NPIX - 1)) + j * 16 + mlane;
      __builtin_prefetch(F + (size_t)pfrow * DCH + bkoff, 0, 1);
#pragma unroll
      for (int kb = 0; kb < 4; ++kb) {
        v16bf bb = ldfragB(F + (size_t)mrow * DCH + kb * 32 + bkoff);
        c0 = wmma_bf16(afr[0][kb], bb, c0);   // B fragment reused twice
        c1 = wmma_bf16(afr[1][kb], bb, c1);
      }
      acc[0][j] = c0;
      acc[1][j] = c1;
    }
    // Online rowmax / sumexp. C layout: row M = v + 8*(lane>=16), col = lane%16,
    // so reductions are butterfly shuffles within each 16-lane half.
#pragma unroll
    for (int h = 0; h < 2; ++h) {
#pragma unroll
      for (int v = 0; v < 8; ++v) {
        float s0 = acc[h][0][v], s1 = acc[h][1][v];
        float bm = fmaxf(s0, s1);
#pragma unroll
        for (int off = 1; off < 16; off <<= 1)
          bm = fmaxf(bm, __shfl_xor(bm, off, 32));
        float nm = fmaxf(rm[h][v], bm);
        float es = __expf(s0 - nm) + __expf(s1 - nm);
#pragma unroll
        for (int off = 1; off < 16; off <<= 1)
          es += __shfl_xor(es, off, 32);
        rs[h][v] = rs[h][v] * __expf(rm[h][v] - nm) + es;
        rm[h][v] = nm;
      }
    }
  }

  if ((lane & 15) == 0) {
    const int half8 = (lane >> 4) << 3;
#pragma unroll
    for (int h = 0; h < 2; ++h) {
#pragma unroll
      for (int v = 0; v < 8; ++v) {
        const int row = r0 + h * 16 + half8 + v;
        statsM[(size_t)b * NPIX + row] = rm[h][v];
        statsS[(size_t)b * NPIX + row] = rs[h][v];
      }
    }
  }
}

// ---------------------------------------------------------------------------
// Pass 2: out[:, m] = sum_n P[n,m] * F[n,:] with P = exp(S - m_n)/l_n.
// Workgroup owns 128 output columns (8 waves x 16). Streams n in blocks of 32.
// ---------------------------------------------------------------------------
__global__ __launch_bounds__(256)
void ca_out_kernel(const __bf16* __restrict__ fb,
                   const float* __restrict__ statsM,
                   const float* __restrict__ statsS,
                   float* __restrict__ out) {
  __shared__ __bf16 lds_fn [32 * DCH];   // F_n block, row-major     (8 KB)
  __shared__ __bf16 lds_fnT[DCH * 32];   // F_n^T (128 x 32)         (8 KB)
  __shared__ __bf16 lds_pT [8][16 * 32]; // per-wave P^T (col x k)   (8 KB)
  __shared__ float  lds_rm[32];
  __shared__ float  lds_rs[32];

  const int b     = blockIdx.y;
  const int lane  = threadIdx.x & 31;
  const int wv    = threadIdx.x >> 5;
  const int mcol0 = blockIdx.x * 128 + wv * 16;
  const __bf16* F = fb + (size_t)b * NPIX * DCH;

  // B fragments for the S GEMM: B = F_m^T restricted to this wave's 16 columns.
  v16bf bfr[4];
  {
    const int mrow = mcol0 + (lane & 15);
    const int koff = (lane & 16) ? 16 : 0;
#pragma unroll
    for (int kb = 0; kb < 4; ++kb)
      bfr[kb] = ldfragB(F + (size_t)mrow * DCH + kb * 32 + koff);
  }

  const v8f vzero = {};
  v8f oacc[8];
#pragma unroll
  for (int t = 0; t < 8; ++t) oacc[t] = vzero;

  const int colL  = lane & 15;
  const int akoff = (lane & 16) ? 16 : 0;

  // Staging assignment: each thread covers a 2-row x 8-col patch so the
  // transposed copy is written with packed b32 stores.
  const int prow = ((int)threadIdx.x >> 4) * 2;       // 0,2,..,30
  const int seg  = ((int)threadIdx.x & 15) * 8;       // 0,8,..,120

  for (int n0 = 0; n0 < NPIX; n0 += 32) {
    // Cooperative stage: F_n (32x128) -> LDS (row-major + transposed), stats.
    {
      const __bf16* s0 = F + (size_t)(n0 + prow)     * DCH + seg;
      const __bf16* s1 = F + (size_t)(n0 + prow + 1) * DCH + seg;
      v8bf a0 = *(const v8bf*)s0;
      v8bf a1 = *(const v8bf*)s1;
      *(v8bf*)(lds_fn + prow * DCH + seg)       = a0;
      *(v8bf*)(lds_fn + (prow + 1) * DCH + seg) = a1;
#pragma unroll
      for (int e = 0; e < 8; ++e) {
        v2bf t; t[0] = a0[e]; t[1] = a1[e];
        *(v2bf*)(lds_fnT + (seg + e) * 32 + prow) = t;
      }
      if (threadIdx.x < 32) {
        lds_rm[threadIdx.x] = statsM[(size_t)b * NPIX + n0 + threadIdx.x];
        lds_rs[threadIdx.x] = statsS[(size_t)b * NPIX + n0 + threadIdx.x];
      }
    }
    __syncthreads();

    // S tile (32 rows x 16 cols) -> P^T staged per wave (one 16B store per mt).
#pragma unroll
    for (int mt = 0; mt < 2; ++mt) {
      v8f acc = vzero;
      const int row = mt * 16 + colL;
#pragma unroll
      for (int kb = 0; kb < 4; ++kb) {
        v16bf a = ldfragA(lds_fn + row * DCH + kb * 32 + akoff);
        acc = wmma_bf16(a, bfr[kb], acc);
      }
      const int nbase = mt * 16 + ((lane & 16) ? 8 : 0);
      v8bf pv;
#pragma unroll
      for (int v = 0; v < 8; ++v) {
        const int nl = nbase + v;
        pv[v] = f2bf(__expf(acc[v] - lds_rm[nl]) / lds_rs[nl]);
      }
      *(v8bf*)(lds_pT[wv] + colL * 32 + nbase) = pv;
    }

    // Reload P as a B operand (K=32 x N=16): contiguous k per lane from P^T.
    const int pk0 = (lane & 16) ? 16 : 0;
    v16bf pb = ldfragB(lds_pT[wv] + colL * 32 + pk0);

    // O(128x16) += F_n^T * P, A fragments from transposed copy (vector loads).
#pragma unroll
    for (int t = 0; t < 8; ++t) {
      const int i = t * 16 + colL;
      v16bf a = ldfragA(lds_fnT + i * 32 + akoff);
      oacc[t] = wmma_bf16(a, pb, oacc[t]);
    }
    __syncthreads();
  }

  // Store O tile: out[b, i, mcol0 + lane%16], i = t*16 + v + 8*(lane>=16).
  {
    const int col = mcol0 + colL;
    const int ib  = (lane & 16) ? 8 : 0;
    float* ob = out + (size_t)b * DCH * NPIX;
#pragma unroll
    for (int t = 0; t < 8; ++t) {
#pragma unroll
      for (int v = 0; v < 8; ++v) {
        ob[(size_t)(t * 16 + ib + v) * NPIX + col] = oacc[t][v];
      }
    }
  }
}

extern "C" void kernel_launch(void* const* d_in, const int* in_sizes, int n_in,
                              void* d_out, int out_size, void* d_ws, size_t ws_size,
                              hipStream_t stream) {
  (void)in_sizes; (void)n_in; (void)out_size; (void)ws_size;
  const float* x = (const float*)d_in[0];
  float* out = (float*)d_out;

  // Workspace layout: bf16 copy of F (8 MB), then softmax stats (256 KB).
  __bf16* Fb = (__bf16*)d_ws;
  float* statsM = (float*)(Fb + (size_t)BATCH * NPIX * DCH);
  float* statsS = statsM + (size_t)BATCH * NPIX;

  const int total4 = (BATCH * NPIX * DCH) / 4;       // float4s
  ca_cvt_kernel<<<total4 / 256, 256, 0, stream>>>(x, Fb);

  dim3 g1(NPIX / 256, BATCH);
  ca_stats_kernel<<<g1, 256, 0, stream>>>(Fb, statsM, statsS);

  dim3 g2(NPIX / 128, BATCH);
  ca_out_kernel<<<g2, 256, 0, stream>>>(Fb, statsM, statsS, out);
}
